// Linear_10634339025298
// MI455X (gfx1250) — 8-bit matrix operand order repaired and checked by reading, compiled, not run on hardware
//
#include <hip/hip_runtime.h>


typedef __attribute__((ext_vector_type(8))) int v8i;
typedef __attribute__((ext_vector_type(4))) int v4i;

#define MT 8192
#define NT 4096
#define KT 4096
#define BM 128
#define BN 128
#define BK 64
#define KSTEPS (KT / BK)

__device__ __forceinline__ unsigned lds_addr32(const void* p) {
  return (unsigned)(uintptr_t)p;
}

__device__ __forceinline__ void async_copy_b128(unsigned lds_off,
                                                const void* gaddr) {
  asm volatile("global_load_async_to_lds_b128 %0, %1, off"
               :
               : "v"(lds_off), "v"(gaddr)
               : "memory");
}

__device__ __forceinline__ void wait_async0() {
  asm volatile("s_wait_asynccnt 0" ::: "memory");
}

__global__ void pack_a_kernel(const int* __restrict__ a32,
                              signed char* __restrict__ a8, long long total) {
  long long idx = ((long long)blockIdx.x * blockDim.x + threadIdx.x) * 4;
  if (idx + 3 < total) {
    v4i v = *(const v4i*)(a32 + idx);
    int packed = (v[0] & 0xff) | ((v[1] & 0xff) << 8) |
                 ((v[2] & 0xff) << 16) | ((v[3] & 0xff) << 24);
    *(int*)(a8 + idx) = packed;
  }
}

__global__ void pack_wt_kernel(const int* __restrict__ w32,
                               signed char* __restrict__ wt8) {
  __shared__ signed char tile[32][33];
  const int n0 = blockIdx.x * 32;
  const int k0 = blockIdx.y * 32;
  const int tx = threadIdx.x & 31;
  const int ty = threadIdx.x >> 5;
#pragma unroll
  for (int r = 0; r < 4; ++r) {
    int k = ty + r * 8;
    tile[k][tx] = (signed char)w32[(size_t)(k0 + k) * NT + (n0 + tx)];
  }
  __syncthreads();
#pragma unroll
  for (int r = 0; r < 4; ++r) {
    int n = ty + r * 8;
    wt8[(size_t)(n0 + n) * KT + (k0 + tx)] = tile[tx][n];
  }
}

__global__ void rowterm_kernel(const int* __restrict__ a32,
                               float* __restrict__ rowterm,
                               const float* __restrict__ a_s_p,
                               const float* __restrict__ b_o_p) {
  const int wave = threadIdx.x >> 5;
  const int lane = threadIdx.x & 31;
  const int row = blockIdx.x * 8 + wave;
  const int* p = a32 + (size_t)row * KT;
  int s = 0;
  for (int k = lane; k < KT; k += 32) s += p[k];
#pragma unroll
  for (int off = 16; off > 0; off >>= 1) s += __shfl_xor(s, off, 32);
  if (lane == 0) rowterm[row] = (*a_s_p) * (*b_o_p) * (float)s;
}

__global__ void colterm_kernel(const int* __restrict__ w32,
                               float* __restrict__ colterm,
                               const float* __restrict__ a_o_p,
                               const float* __restrict__ b_s_p) {
  const int n = blockIdx.x * blockDim.x + threadIdx.x;
  int s = 0;
  for (int k = 0; k < KT; ++k) s += w32[(size_t)k * NT + n];
  colterm[n] = (*a_o_p) * (*b_s_p) * (float)s;
}

__global__ __launch_bounds__(256)
void gemm_i8_dequant_kernel(const signed char* __restrict__ A,
                            const signed char* __restrict__ WT,
                            const float* __restrict__ rowterm,
                            const float* __restrict__ colterm,
                            const float* __restrict__ a_s_p,
                            const float* __restrict__ a_o_p,
                            const float* __restrict__ b_s_p,
                            const float* __restrict__ b_o_p,
                            float* __restrict__ out) {
  __shared__ __align__(16) signed char lA[2][BM][BK];
  __shared__ __align__(16) signed char lB[2][BN][BK];

  const int m0 = blockIdx.y * BM;
  const int n0 = blockIdx.x * BN;
  const int t = threadIdx.x;
  const int lane = t & 31;
  const int wave = t >> 5;
  const int wave_m = wave & 1;
  const int wave_n = wave >> 1;

  const v8i vzero = {0, 0, 0, 0, 0, 0, 0, 0};
  v8i acc[4][2];
#pragma unroll
  for (int i = 0; i < 4; ++i)
#pragma unroll
    for (int j = 0; j < 2; ++j) acc[i][j] = vzero;

  auto load_tiles_async = [&](int buf, int kt) {
    const int k0 = kt * BK;
#pragma unroll
    for (int h = 0; h < 2; ++h) {
      const int c = t + h * 256;
      const int row = c >> 2;
      const int seg = (c & 3) * 16;
      const signed char* ga = A + (size_t)(m0 + row) * KT + k0 + seg;
      const signed char* gb = WT + (size_t)(n0 + row) * KT + k0 + seg;
      async_copy_b128(lds_addr32(&lA[buf][row][seg]), ga);
      async_copy_b128(lds_addr32(&lB[buf][row][seg]), gb);
      __builtin_prefetch(ga + 2 * BK, 0, 1);
      __builtin_prefetch(gb + 2 * BK, 0, 1);
    }
  };

  auto compute = [&](int buf) {
    const int rsel = lane >> 4;
    const int lrow = lane & 15;
    v8i af[4], bf[2];
#pragma unroll
    for (int i = 0; i < 4; ++i) {
      const signed char* ap = &lA[buf][wave_m * 64 + i * 16 + lrow][0];
      union { unsigned long long q[4]; v8i v; } u;
#pragma unroll
      for (int j = 0; j < 4; ++j)
        u.q[j] = *(const unsigned long long*)(ap + (j >> 1) * 32 + rsel * 16 + (j & 1) * 8);
      af[i] = u.v;
    }
#pragma unroll
    for (int j = 0; j < 2; ++j) {
      const signed char* bp = &lB[buf][wave_n * 32 + j * 16 + lrow][0];
      union { v4i x[2]; v8i v; } u;
      u.x[0] = *(const v4i*)(bp + rsel * 16);
      u.x[1] = *(const v4i*)(bp + 32 + rsel * 16);
      bf[j] = u.v;
    }
#pragma unroll
    for (int i = 0; i < 4; ++i)
#pragma unroll
      for (int j = 0; j < 2; ++j)
        acc[i][j] = __builtin_amdgcn_wmma_i32_16x16x64_iu8(
             true, af[i],  true, bf[j], acc[i][j],
             false,  false);
  };

  load_tiles_async(0, 0);
  wait_async0();
  __syncthreads();
#pragma unroll 2
  for (int kt = 0; kt < KSTEPS; ++kt) {
    const int buf = kt & 1;
    if (kt + 1 < KSTEPS) load_tiles_async(buf ^ 1, kt + 1);
    compute(buf);
    wait_async0();
    __syncthreads();
  }

  const float a_s = *a_s_p, a_o = *a_o_p, b_s = *b_s_p, b_o = *b_o_p;
  const float fscale = a_s * b_s;
  const float kconst = (float)KT * a_o * b_o;
  const int rsel8 = (lane >> 4) * 8;
  const int lcol = lane & 15;
#pragma unroll
  for (int i = 0; i < 4; ++i) {
    const int mb = m0 + wave_m * 64 + i * 16;
    float rt[8];
#pragma unroll
    for (int v = 0; v < 8; ++v) rt[v] = rowterm[mb + rsel8 + v];
#pragma unroll
    for (int j = 0; j < 2; ++j) {
      const int gn = n0 + wave_n * 32 + j * 16 + lcol;
      const float ct = colterm[gn] + kconst;
#pragma unroll
      for (int v = 0; v < 8; ++v)
        out[(size_t)(mb + rsel8 + v) * NT + gn] =
            (float)acc[i][j][v] * fscale + rt[v] + ct;
    }
  }
}

extern "C" void kernel_launch(void* const* d_in, const int* in_sizes, int n_in,
                              void* d_out, int out_size, void* d_ws,
                              size_t ws_size, hipStream_t stream) {
  const int* a32 = (const int*)d_in[0];
  const int* w32 = (const int*)d_in[1];
  const float* a_s = (const float*)d_in[2];
  const float* a_o = (const float*)d_in[3];
  const float* b_s = (const float*)d_in[4];
  const float* b_o = (const float*)d_in[5];
  float* out = (float*)d_out;

  signed char* a8 = (signed char*)d_ws;
  signed char* wt8 = a8 + (size_t)MT * KT;
  float* rowterm = (float*)(wt8 + (size_t)NT * KT);
  float* colterm = rowterm + MT;

  const long long a_total = (long long)MT * KT;
  pack_a_kernel<<<(unsigned)(a_total / 4 / 256), 256, 0, stream>>>(a32, a8, a_total);
  pack_wt_kernel<<<dim3(NT / 32, KT / 32), 256, 0, stream>>>(w32, wt8);
  rowterm_kernel<<<MT / 8, 256, 0, stream>>>(a32, rowterm, a_s, b_o);
  colterm_kernel<<<NT / 256, 256, 0, stream>>>(w32, colterm, a_o, b_s);
  gemm_i8_dequant_kernel<<<dim3(NT / BN, MT / BM), 256, 0, stream>>>(
      a8, wt8, rowterm, colterm, a_s, a_o, b_s, b_o, out);
}
